// EvoformerBlock_62998580297867
// MI455X (gfx1250) — compile-verified, matrix operand order checked by reading, not run on hardware
//
#include <hip/hip_runtime.h>

// ---------------------------------------------------------------------------
// Evoformer block for MI455X (gfx1250): bf16 WMMA everywhere, fp32 accumulate.
// v2: 64x64 register-blocked WMMA GEMM (16 wmma / K-step), branch-free hot
//     loop, global_prefetch of the K panel after next.
// ---------------------------------------------------------------------------

#define DEVFN static __device__ __forceinline__

typedef __attribute__((ext_vector_type(16))) __bf16 bvec16;
typedef __attribute__((ext_vector_type(8)))  float  fvec8;

struct u128 { unsigned int a, b, c, d; };
union ABu { bvec16 v; u128 q[2]; unsigned short u[16]; };

DEVFN unsigned short f2b(float f) {
  unsigned int u = __float_as_uint(f);
  return (unsigned short)((u + 0x7FFFu + ((u >> 16) & 1u)) >> 16);
}
DEVFN float b2f(unsigned short h) { return __uint_as_float(((unsigned int)h) << 16); }
DEVFN int kmap(int j, int kb) { return (j < 8) ? (kb + j) : (kb + 8 + j); }
DEVFN fvec8 wmma_bf(const ABu& a, const ABu& b, fvec8 c) {
  return __builtin_amdgcn_wmma_f32_16x16x32_bf16(false, a.v, false, b.v, (short)0, c, false, false);
}

// ---- problem dims ---------------------------------------------------------
static constexpr int  Nn = 128, Ll = 192, CM = 256, CZ = 128;
static constexpr int  Hh = 8, DHc = 32, HT = 4, CTRI = 128;
static constexpr long SM = (long)Nn * Ll;   // 24576 MSA positions
static constexpr long SZ = (long)Ll * Ll;   // 36864 pair positions

// ---------------------------------------------------------------------------
// Kernels
// ---------------------------------------------------------------------------

// f32 (K,N) weight -> bf16 transposed (N,K)
__global__ void convw_kernel(const float* __restrict__ w, unsigned short* __restrict__ wt,
                             int K, int Nc) {
  long i = (long)blockIdx.x * 256 + threadIdx.x;
  long tot = (long)K * Nc;
  if (i >= tot) return;
  long n = i / K, k = i - n * K;
  wt[i] = f2b(w[k * (long)Nc + n]);
}

// LayerNorm over last dim C, one wave per row, bf16 out.
__global__ void __launch_bounds__(32) ln_kernel(
    const float* __restrict__ x, const float* __restrict__ g,
    const float* __restrict__ b, unsigned short* __restrict__ out, int C) {
  long row = blockIdx.x;
  const float* xr = x + row * (long)C;
  int lane = threadIdx.x;
  float s = 0.f;
  for (int c = lane; c < C; c += 32) s += xr[c];
  for (int o = 16; o; o >>= 1) s += __shfl_xor(s, o, 32);
  float mu = s / C;
  float v = 0.f;
  for (int c = lane; c < C; c += 32) { float d = xr[c] - mu; v += d * d; }
  for (int o = 16; o; o >>= 1) v += __shfl_xor(v, o, 32);
  float inv = rsqrtf(v / C + 1e-5f);
  unsigned short* orow = out + row * (long)C;
  for (int c = lane; c < C; c += 32) orow[c] = f2b((xr[c] - mu) * inv * g[c] + b[c]);
}

// Generic bf16 GEMM, 64x64 tile per wave: out = act(A @ BT^T + bias) [*gate] [+= old]
struct GemmP {
  const unsigned short* A;    // (M,K) row-major bf16
  const unsigned short* BT;   // (N,K) row-major bf16 (weight transposed)
  const float* bias;          // [Nc] or null
  const float* gate;          // f32 same shape as out, or null
  float* outf;                // f32 out or null
  unsigned short* outb;       // bf16 out or null
  int M, Nc, K, lda, ldbt, ldc, act, accum;   // act: 0 none, 1 relu, 2 sigmoid
};
__global__ void __launch_bounds__(32) gemm_kernel(GemmP p) {
  int lane = threadIdx.x;
  int n0 = blockIdx.x * 64, m0 = blockIdx.y * 64;
  int kb = (lane & 16) ? 8 : 0;
  int arow = m0 + (lane & 15);
  int brow = n0 + (lane & 15);
  const bool full = (n0 + 64 <= p.Nc);
  const unsigned short* Abase = p.A + (long)arow * p.lda + kb;
  const unsigned short* Bbase = p.BT + (long)brow * p.ldbt + kb;
  const long astep = (long)16 * p.lda;
  const long bstep = (long)16 * p.ldbt;
  fvec8 acc[4][4] = {};
  for (int k0 = 0; k0 < p.K; k0 += 32) {
    ABu af[4], bf[4];
#pragma unroll
    for (int t = 0; t < 4; ++t) {
      const u128* pa = (const u128*)(Abase + (long)t * astep + k0);
      af[t].q[0] = pa[0]; af[t].q[1] = pa[2];
    }
    if (full) {
#pragma unroll
      for (int t = 0; t < 4; ++t) {
        const u128* pb = (const u128*)(Bbase + (long)t * bstep + k0);
        bf[t].q[0] = pb[0]; bf[t].q[1] = pb[2];
      }
    } else {
#pragma unroll
      for (int t = 0; t < 4; ++t) {
        if (brow + t * 16 < p.Nc) {
          const u128* pb = (const u128*)(Bbase + (long)t * bstep + k0);
          bf[t].q[0] = pb[0]; bf[t].q[1] = pb[2];
        } else {
          u128 zz = {0u, 0u, 0u, 0u}; bf[t].q[0] = zz; bf[t].q[1] = zz;
        }
      }
    }
    if (k0 + 64 < p.K) {  // pull the K panel after next into cache
      __builtin_prefetch((const void*)(Abase + k0 + 64), 0, 1);
      __builtin_prefetch((const void*)(Bbase + k0 + 64), 0, 1);
    }
#pragma unroll
    for (int mi = 0; mi < 4; ++mi)
#pragma unroll
      for (int ni = 0; ni < 4; ++ni)
        acc[mi][ni] = wmma_bf(af[mi], bf[ni], acc[mi][ni]);
  }
  int rbase = (lane & 16) ? 8 : 0;
#pragma unroll
  for (int ni = 0; ni < 4; ++ni) {
    int col = n0 + ni * 16 + (lane & 15);
    if (col >= p.Nc) continue;
    float bias = p.bias ? p.bias[col] : 0.f;
#pragma unroll
    for (int mi = 0; mi < 4; ++mi) {
#pragma unroll
      for (int vi = 0; vi < 8; ++vi) {
        int r = m0 + mi * 16 + rbase + vi;
        long idx = (long)r * p.ldc + col;
        float val = acc[mi][ni][vi] + bias;
        if (p.act == 1) val = fmaxf(val, 0.f);
        else if (p.act == 2) val = 1.f / (1.f + __expf(-val));
        if (p.gate) val *= p.gate[idx];
        if (p.outf) p.outf[idx] = p.accum ? (p.outf[idx] + val) : val;
        if (p.outb) p.outb[idx] = f2b(val);
      }
    }
  }
}

// Fused attention: one wave per (batch, head, 16-query tile). DH fixed at 32.
struct AttnP {
  const unsigned short *q, *k, *v;  // layout: base + b*q_bs + row*q_rs + h*32 + d
  const float* bias;                // bias[h + row*b_rs + key*b_cs] or null
  float* out;                       // same layout as q
  long q_bs; int q_rs;
  long b_rs; int b_cs;
  int Q, S;
  float scale;
};
__global__ void __launch_bounds__(32) attn_kernel(AttnP p) {
  __shared__ float Lg[16][200];
  __shared__ unsigned short Pb[16][192];
  __shared__ float rmax[16], rsum[16];
  int lane = threadIdx.x;
  int h = blockIdx.y;
  long base = (long)blockIdx.z * p.q_bs + (long)h * 32;
  int q0 = blockIdx.x * 16;
  int kb = (lane & 16) ? 8 : 0;

  ABu qf;
  {
    int row = q0 + (lane & 15);
    const u128* ptr = (const u128*)(p.q + base + (long)row * p.q_rs + kb);
    qf.q[0] = ptr[0]; qf.q[1] = ptr[2];
  }
  // logits = q @ k^T * scale + bias
  for (int sc = 0; sc < p.S / 16; ++sc) {
    ABu kf;
    int krow = sc * 16 + (lane & 15);
    const u128* ptr = (const u128*)(p.k + base + (long)krow * p.q_rs + kb);
    kf.q[0] = ptr[0]; kf.q[1] = ptr[2];
    fvec8 c = {};
    c = wmma_bf(qf, kf, c);
    int col = sc * 16 + (lane & 15);
    int rbase = (lane & 16) ? 8 : 0;
#pragma unroll
    for (int vi = 0; vi < 8; ++vi) {
      int r = rbase + vi;
      float val = c[vi] * p.scale;
      if (p.bias) val += p.bias[h + (long)(q0 + r) * p.b_rs + (long)col * p.b_cs];
      Lg[r][col] = val;
    }
  }
  __syncthreads();
  // row softmax (2 lanes per row)
  {
    int r = lane >> 1, half = lane & 1;
    int c0 = half * (p.S / 2), c1 = c0 + p.S / 2;
    float mx = -1e30f;
    for (int c = c0; c < c1; ++c) mx = fmaxf(mx, Lg[r][c]);
    mx = fmaxf(mx, __shfl_xor(mx, 1, 32));
    float sum = 0.f;
    for (int c = c0; c < c1; ++c) sum += __expf(Lg[r][c] - mx);
    sum += __shfl_xor(sum, 1, 32);
    if (half == 0) { rmax[r] = mx; rsum[r] = sum; }
  }
  __syncthreads();
  for (int idx = lane; idx < 16 * p.S; idx += 32) {
    int r = idx / p.S, c = idx - r * p.S;
    Pb[r][c] = f2b(__expf(Lg[r][c] - rmax[r]) / rsum[r]);
  }
  __syncthreads();
  // o = P @ V
  fvec8 o0 = {}, o1 = {};
  for (int kc = 0; kc < p.S / 32; ++kc) {
    ABu pf, vf0, vf1;
    int row = lane & 15;
    int n = lane & 15;
#pragma unroll
    for (int j = 0; j < 16; ++j) {
      int kk = kc * 32 + kmap(j, kb);
      pf.u[j] = Pb[row][kk];
      const unsigned short* vp = p.v + base + (long)kk * p.q_rs;
      vf0.u[j] = vp[n];
      vf1.u[j] = vp[n + 16];
    }
    o0 = wmma_bf(pf, vf0, o0);
    o1 = wmma_bf(pf, vf1, o1);
  }
  {
    int colb = lane & 15;
    int rbase = (lane & 16) ? 8 : 0;
#pragma unroll
    for (int vi = 0; vi < 8; ++vi) {
      int r = q0 + rbase + vi;
      float* op = p.out + base + (long)r * p.q_rs;
      op[colb] = o0[vi];
      op[colb + 16] = o1[vi];
    }
  }
}

// Outer product mean pairs: out[(i*L+j)*1024 + c*32+d] = (1/Nr) sum_n a[n,i,c]*b[n,j,d]
__global__ void __launch_bounds__(32) opm_pair_kernel(
    const unsigned short* __restrict__ a, const unsigned short* __restrict__ bm,
    unsigned short* __restrict__ out, int Lr, int Nr) {
  int i = blockIdx.x, j = blockIdx.y;
  int lane = threadIdx.x;
  int kb = (lane & 16) ? 8 : 0;
  fvec8 acc[2][2] = {};
  for (int kc = 0; kc < Nr; kc += 32) {
    ABu af[2], bf[2];
#pragma unroll
    for (int mi = 0; mi < 2; ++mi) {
      int c = (lane & 15) + mi * 16;
#pragma unroll
      for (int jj = 0; jj < 16; ++jj) {
        int n = kc + kmap(jj, kb);
        af[mi].u[jj] = a[((long)n * Lr + i) * 32 + c];
        bf[mi].u[jj] = bm[((long)n * Lr + j) * 32 + c];
      }
    }
#pragma unroll
    for (int mi = 0; mi < 2; ++mi)
#pragma unroll
      for (int ni = 0; ni < 2; ++ni)
        acc[mi][ni] = wmma_bf(af[mi], bf[ni], acc[mi][ni]);
  }
  float sc = 1.f / (float)Nr;
  unsigned short* orow = out + ((long)i * Lr + j) * 1024;
  int rbase = (lane & 16) ? 8 : 0;
#pragma unroll
  for (int mi = 0; mi < 2; ++mi)
#pragma unroll
    for (int ni = 0; ni < 2; ++ni)
#pragma unroll
      for (int vi = 0; vi < 8; ++vi) {
        int c = rbase + vi + mi * 16;
        int d = (lane & 15) + ni * 16;
        orow[c * 32 + d] = f2b(acc[mi][ni][vi] * sc);
      }
}

// Triangle-mul einsum: x[i,j,c] = sum_k a[.]*b[.] (strides select outgoing/incoming)
struct TriP {
  const unsigned short* a; const unsigned short* b; float* x;
  long a_is, a_ks, b_js, b_ks;
  int Lr, C, K;
};
__global__ void __launch_bounds__(32) trimul_kernel(TriP p) {
  int i0 = blockIdx.x * 16, j0 = blockIdx.y * 16, c = blockIdx.z;
  int lane = threadIdx.x;
  int kb = (lane & 16) ? 8 : 0;
  int ir = i0 + (lane & 15), jr = j0 + (lane & 15);
  fvec8 acc = {};
  for (int kc = 0; kc < p.K; kc += 32) {
    ABu af, bf;
#pragma unroll
    for (int jj = 0; jj < 16; ++jj) {
      int k = kc + kmap(jj, kb);
      af.u[jj] = p.a[(long)ir * p.a_is + (long)k * p.a_ks + c];
      bf.u[jj] = p.b[(long)jr * p.b_js + (long)k * p.b_ks + c];
    }
    acc = wmma_bf(af, bf, acc);
  }
  int j = j0 + (lane & 15);
  int rbase = (lane & 16) ? 8 : 0;
#pragma unroll
  for (int vi = 0; vi < 8; ++vi) {
    int i = i0 + rbase + vi;
    p.x[((long)i * p.Lr + j) * p.C + c] = acc[vi];
  }
}

// Elementwise helpers
__global__ void mul_bf_f32_kernel(const unsigned short* a, const float* b,
                                  unsigned short* out, long n) {
  long i = (long)blockIdx.x * 256 + threadIdx.x;
  if (i < n) out[i] = f2b(b2f(a[i]) * b[i]);
}
__global__ void mul_ff_bf_kernel(const float* a, const float* b, unsigned short* out, long n) {
  long i = (long)blockIdx.x * 256 + threadIdx.x;
  if (i < n) out[i] = f2b(a[i] * b[i]);
}
__global__ void transpose_bf_kernel(const unsigned short* in, unsigned short* out, int Lr, int C) {
  long i = (long)blockIdx.x * 256 + threadIdx.x;
  long n = (long)Lr * Lr * C;
  if (i >= n) return;
  int c = (int)(i % C);
  long t = i / C;
  int jj = (int)(t % Lr), ii = (int)(t / Lr);
  out[i] = in[((long)jj * Lr + ii) * C + c];
}
__global__ void addT_kernel(float* z, const float* src, int Lr, int C) {
  long i = (long)blockIdx.x * 256 + threadIdx.x;
  long n = (long)Lr * Lr * C;
  if (i >= n) return;
  int c = (int)(i % C);
  long t = i / C;
  int jj = (int)(t % Lr), ii = (int)(t / Lr);
  z[i] += src[((long)jj * Lr + ii) * C + c];
}

// ---------------------------------------------------------------------------
// Host orchestration
// ---------------------------------------------------------------------------
static constexpr size_t AL(size_t x) { return (x + 255) & ~(size_t)255; }
static constexpr size_t SZ_W    = (size_t)8 << 20;
static constexpr size_t OFF_W   = 0;
static constexpr size_t OFF_LNA = OFF_W + SZ_W;                   // SM*CM bf16 (mn / xn)
static constexpr size_t OFF_LNB = OFF_LNA + AL((size_t)SM*CM*2);  // SZ*CZ bf16 (zn)
static constexpr size_t OFF_Q   = OFF_LNB + AL((size_t)SZ*CZ*2);
static constexpr size_t OFF_K   = OFF_Q  + AL((size_t)SM*CM*2);
static constexpr size_t OFF_V   = OFF_K  + AL((size_t)SM*CM*2);
static constexpr size_t OFF_G   = OFF_V  + AL((size_t)SM*CM*2);
static constexpr size_t OFF_BIA = OFF_G  + AL((size_t)SM*CM*2);   // SZ*8 f32
static constexpr size_t OFF_ATO = OFF_BIA + AL((size_t)SZ*8*4);   // SM*CM f32
static constexpr size_t OFF_GO  = OFF_ATO + AL((size_t)SM*CM*4);  // SM*CM bf16
static constexpr size_t OFF_T1  = OFF_GO  + AL((size_t)SM*CM*2);  // SZ*CTRI f32
static constexpr size_t OFF_T2  = OFF_T1  + AL((size_t)SZ*CTRI*4);
static constexpr size_t OFF_ABF = OFF_T2  + AL((size_t)SZ*CTRI*4);
static constexpr size_t OFF_BBF = OFF_ABF + AL((size_t)SZ*CTRI*2);
static constexpr size_t OFF_X   = OFF_BBF + AL((size_t)SZ*CTRI*2);
static constexpr size_t OFF_ZNT = OFF_X   + AL((size_t)SZ*CTRI*4);
static constexpr size_t OFF_HID = OFF_ZNT + AL((size_t)SZ*CZ*2);  // SZ*1024 bf16 arena
static constexpr size_t WS_NEED = OFF_HID + AL((size_t)SZ*1024*2);

extern "C" void kernel_launch(void* const* d_in, const int* in_sizes, int n_in,
                              void* d_out, int out_size, void* d_ws, size_t ws_size,
                              hipStream_t stream) {
  if (n_in < 95 || ws_size < WS_NEED) return;
  // Detect params flattening: insertion order -> in_sizes[4]==65536 (msa_row.wq),
  // jax-sorted order -> in_sizes[4]==256 (msa_col.ln.b).
  const bool so = (in_sizes[4] == 256);
  auto PIN = [&](int ins, int srt) -> const float* {
    return (const float*)d_in[so ? srt : ins];
  };
  char* ws = (char*)d_ws;
  auto WSb = [&](size_t off) { return (unsigned short*)(ws + off); };
  auto WSf = [&](size_t off) { return (float*)(ws + off); };

  float* m = (float*)d_out;
  float* z = m + SM * CM;
  unsigned short* mn   = WSb(OFF_LNA);
  unsigned short* zn   = WSb(OFF_LNB);
  unsigned short* qb   = WSb(OFF_Q);
  unsigned short* kb   = WSb(OFF_K);
  unsigned short* vb   = WSb(OFF_V);
  unsigned short* gb   = WSb(OFF_G);
  float*          bia  = WSf(OFF_BIA);
  float*          ato  = WSf(OFF_ATO);
  unsigned short* go   = WSb(OFF_GO);
  float*          t1   = WSf(OFF_T1);
  float*          t2   = WSf(OFF_T2);
  unsigned short* abf  = WSb(OFF_ABF);
  unsigned short* bbf  = WSb(OFF_BBF);
  float*          xf   = WSf(OFF_X);
  unsigned short* znT  = WSb(OFF_ZNT);
  unsigned short* hid  = WSb(OFF_HID);

  // ---- weight conversion (f32 (K,N) -> bf16 (N,K)) ----
  size_t wcur = 0;
  auto CONV = [&](const float* w, int K, int Nc) -> unsigned short* {
    unsigned short* wt = (unsigned short*)(ws + OFF_W + wcur);
    wcur += AL((size_t)K * Nc * 2);
    long tot = (long)K * Nc;
    convw_kernel<<<dim3((unsigned)((tot + 255) / 256)), 256, 0, stream>>>(w, wt, K, Nc);
    return wt;
  };
  // msa_row
  unsigned short* wq_r = CONV(PIN(4, 21), 256, 256);
  unsigned short* wk_r = CONV(PIN(5, 19), 256, 256);
  unsigned short* wv_r = CONV(PIN(6, 22), 256, 256);
  unsigned short* wg_r = CONV(PIN(7, 18), 256, 256);
  unsigned short* wo_r = CONV(PIN(9, 20), 256, 256);
  unsigned short* wb_r = CONV(PIN(11, 17), 128, 8);
  // msa_col
  unsigned short* wq_c = CONV(PIN(16, 9), 256, 256);
  unsigned short* wk_c = CONV(PIN(17, 7), 256, 256);
  unsigned short* wv_c = CONV(PIN(18, 10), 256, 256);
  unsigned short* wg_c = CONV(PIN(19, 6), 256, 256);
  unsigned short* wo_c = CONV(PIN(21, 8), 256, 256);
  // msa_trans
  unsigned short* w1_m = CONV(PIN(25, 27), 256, 1024);
  unsigned short* w2_m = CONV(PIN(27, 28), 1024, 256);
  // opm
  unsigned short* wa_o = CONV(PIN(31, 34), 256, 32);
  unsigned short* wb_o = CONV(PIN(33, 35), 256, 32);
  unsigned short* wo_o = CONV(PIN(35, 36), 1024, 128);
  // tri_out
  unsigned short* wap_to = CONV(PIN(39, 80), 128, 128);
  unsigned short* wag_to = CONV(PIN(41, 79), 128, 128);
  unsigned short* wbp_to = CONV(PIN(43, 82), 128, 128);
  unsigned short* wbg_to = CONV(PIN(45, 81), 128, 128);
  unsigned short* wg_to  = CONV(PIN(49, 83), 128, 128);
  unsigned short* wo_to  = CONV(PIN(51, 84), 128, 128);
  // tri_in
  unsigned short* wap_ti = CONV(PIN(55, 64), 128, 128);
  unsigned short* wag_ti = CONV(PIN(57, 63), 128, 128);
  unsigned short* wbp_ti = CONV(PIN(59, 66), 128, 128);
  unsigned short* wbg_ti = CONV(PIN(61, 65), 128, 128);
  unsigned short* wg_ti  = CONV(PIN(65, 67), 128, 128);
  unsigned short* wo_ti  = CONV(PIN(67, 68), 128, 128);
  // tri_start
  unsigned short* wq_ts = CONV(PIN(71, 93), 128, 128);
  unsigned short* wk_ts = CONV(PIN(72, 91), 128, 128);
  unsigned short* wv_ts = CONV(PIN(73, 94), 128, 128);
  unsigned short* wg_ts = CONV(PIN(74, 90), 128, 128);
  unsigned short* wo_ts = CONV(PIN(76, 92), 128, 128);
  unsigned short* wb_ts = CONV(PIN(78, 89), 128, 4);
  // tri_end
  unsigned short* wq_te = CONV(PIN(81, 51), 128, 128);
  unsigned short* wk_te = CONV(PIN(82, 49), 128, 128);
  unsigned short* wv_te = CONV(PIN(83, 52), 128, 128);
  unsigned short* wg_te = CONV(PIN(84, 48), 128, 128);
  unsigned short* wo_te = CONV(PIN(86, 50), 128, 128);
  unsigned short* wb_te = CONV(PIN(88, 47), 128, 4);
  // pair_trans
  unsigned short* w1_p = CONV(PIN(91, 41), 128, 512);
  unsigned short* w2_p = CONV(PIN(93, 42), 512, 128);

  auto GEMM = [&](const unsigned short* A, const unsigned short* BT, const float* bias,
                  long M, int Nc, int Kk, int act, float* outf, unsigned short* outb,
                  const float* gate, int accum) {
    GemmP p; p.A = A; p.BT = BT; p.bias = bias; p.gate = gate; p.outf = outf; p.outb = outb;
    p.M = (int)M; p.Nc = Nc; p.K = Kk; p.lda = Kk; p.ldbt = Kk; p.ldc = Nc;
    p.act = act; p.accum = accum;
    gemm_kernel<<<dim3((unsigned)((Nc + 63) / 64), (unsigned)(M / 64)), 32, 0, stream>>>(p);
  };
  auto LN = [&](const float* x, const float* g, const float* b, unsigned short* out,
                long rows, int C) {
    ln_kernel<<<dim3((unsigned)rows), 32, 0, stream>>>(x, g, b, out, C);
  };
  auto EW = [&](long n) { return dim3((unsigned)((n + 255) / 256)); };
  const float SCALE = 0.17677669529663689f;  // 1/sqrt(32)

  // ---- init residual streams in d_out ----
  hipMemcpyAsync(m, d_in[0], (size_t)SM * CM * 4, hipMemcpyDeviceToDevice, stream);
  hipMemcpyAsync(z, d_in[1], (size_t)SZ * CZ * 4, hipMemcpyDeviceToDevice, stream);

  // ================= MSA row attention (pair bias) =================
  LN(m, PIN(2, 14), PIN(3, 13), mn, SM, CM);
  LN(z, PIN(12, 16), PIN(13, 15), zn, SZ, CZ);
  GEMM(mn, wq_r, nullptr, SM, 256, 256, 0, nullptr, qb, nullptr, 0);
  GEMM(mn, wk_r, nullptr, SM, 256, 256, 0, nullptr, kb, nullptr, 0);
  GEMM(mn, wv_r, nullptr, SM, 256, 256, 0, nullptr, vb, nullptr, 0);
  GEMM(mn, wg_r, PIN(8, 11), SM, 256, 256, 2, nullptr, gb, nullptr, 0);
  GEMM(zn, wb_r, nullptr, SZ, 8, 128, 0, bia, nullptr, nullptr, 0);
  {
    AttnP p; p.q = qb; p.k = kb; p.v = vb; p.bias = bia; p.out = ato;
    p.q_bs = (long)Ll * CM; p.q_rs = CM;
    p.b_rs = (long)Ll * Hh; p.b_cs = Hh;
    p.Q = Ll; p.S = Ll; p.scale = SCALE;
    attn_kernel<<<dim3(Ll / 16, Hh, Nn), 32, 0, stream>>>(p);
  }
  mul_bf_f32_kernel<<<EW(SM * CM), 256, 0, stream>>>(gb, ato, go, SM * CM);
  GEMM(go, wo_r, PIN(10, 12), SM, 256, 256, 0, m, nullptr, nullptr, 1);

  // ================= MSA column attention =================
  LN(m, PIN(14, 5), PIN(15, 4), mn, SM, CM);
  GEMM(mn, wq_c, nullptr, SM, 256, 256, 0, nullptr, qb, nullptr, 0);
  GEMM(mn, wk_c, nullptr, SM, 256, 256, 0, nullptr, kb, nullptr, 0);
  GEMM(mn, wv_c, nullptr, SM, 256, 256, 0, nullptr, vb, nullptr, 0);
  GEMM(mn, wg_c, PIN(20, 2), SM, 256, 256, 2, nullptr, gb, nullptr, 0);
  {
    AttnP p; p.q = qb; p.k = kb; p.v = vb; p.bias = nullptr; p.out = ato;
    p.q_bs = (long)CM; p.q_rs = Ll * CM;   // batch = column l, rows = MSA index n
    p.b_rs = 0; p.b_cs = 0;
    p.Q = Nn; p.S = Nn; p.scale = SCALE;
    attn_kernel<<<dim3(Nn / 16, Hh, Ll), 32, 0, stream>>>(p);
  }
  mul_bf_f32_kernel<<<EW(SM * CM), 256, 0, stream>>>(gb, ato, go, SM * CM);
  GEMM(go, wo_c, PIN(22, 3), SM, 256, 256, 0, m, nullptr, nullptr, 1);

  // ================= MSA transition =================
  LN(m, PIN(23, 26), PIN(24, 25), mn, SM, CM);
  GEMM(mn, w1_m, PIN(26, 23), SM, 1024, 256, 1, nullptr, hid, nullptr, 0);
  GEMM(hid, w2_m, PIN(28, 24), SM, 256, 1024, 0, m, nullptr, nullptr, 1);

  // ================= Outer product mean =================
  LN(m, PIN(29, 33), PIN(30, 32), mn, SM, CM);
  GEMM(mn, wa_o, PIN(32, 29), SM, 32, 256, 0, nullptr, abf, nullptr, 0);
  GEMM(mn, wb_o, PIN(34, 30), SM, 32, 256, 0, nullptr, bbf, nullptr, 0);
  opm_pair_kernel<<<dim3(Ll, Ll), 32, 0, stream>>>(abf, bbf, hid, Ll, Nn);
  GEMM(hid, wo_o, PIN(36, 31), SZ, 128, 1024, 0, z, nullptr, nullptr, 1);

  // ================= Triangle multiplicative (outgoing / incoming) =================
  struct TriW { const unsigned short *wap,*wag,*wbp,*wbg,*wg,*wo;
                int lg_i,lg_s, lb_i,lb_s, bap_i,bap_s, bag_i,bag_s,
                    bbp_i,bbp_s, bbg_i,bbg_s, log_i,log_s, lob_i,lob_s,
                    bg_i,bg_s, bo_i,bo_s; bool outgoing; };
  TriW trs[2] = {
    { wap_to, wag_to, wbp_to, wbg_to, wg_to, wo_to,
      37,76, 38,75, 40,70, 42,69, 44,72, 46,71, 47,78, 48,77, 50,73, 52,74, true },
    { wap_ti, wag_ti, wbp_ti, wbg_ti, wg_ti, wo_ti,
      53,60, 54,59, 56,54, 58,53, 60,56, 62,55, 63,62, 64,61, 66,57, 68,58, false },
  };
  for (int t = 0; t < 2; ++t) {
    TriW& w = trs[t];
    LN(z, PIN(w.lg_i, w.lg_s), PIN(w.lb_i, w.lb_s), zn, SZ, CZ);
    GEMM(zn, w.wap, PIN(w.bap_i, w.bap_s), SZ, 128, 128, 0, t1, nullptr, nullptr, 0);
    GEMM(zn, w.wag, PIN(w.bag_i, w.bag_s), SZ, 128, 128, 2, t2, nullptr, nullptr, 0);
    mul_ff_bf_kernel<<<EW(SZ * 128), 256, 0, stream>>>(t2, t1, abf, SZ * 128);
    GEMM(zn, w.wbp, PIN(w.bbp_i, w.bbp_s), SZ, 128, 128, 0, t1, nullptr, nullptr, 0);
    GEMM(zn, w.wbg, PIN(w.bbg_i, w.bbg_s), SZ, 128, 128, 2, t2, nullptr, nullptr, 0);
    mul_ff_bf_kernel<<<EW(SZ * 128), 256, 0, stream>>>(t2, t1, bbf, SZ * 128);
    GEMM(zn, w.wg, PIN(w.bg_i, w.bg_s), SZ, 128, 128, 2, t1, nullptr, nullptr, 0);
    TriP tp; tp.a = abf; tp.b = bbf; tp.x = xf; tp.Lr = Ll; tp.C = CTRI; tp.K = Ll;
    if (w.outgoing) { tp.a_is = (long)Ll * CTRI; tp.a_ks = CTRI;
                      tp.b_js = (long)Ll * CTRI; tp.b_ks = CTRI; }
    else            { tp.a_is = CTRI; tp.a_ks = (long)Ll * CTRI;
                      tp.b_js = CTRI; tp.b_ks = (long)Ll * CTRI; }
    trimul_kernel<<<dim3(Ll / 16, Ll / 16, CTRI), 32, 0, stream>>>(tp);
    LN(xf, PIN(w.log_i, w.log_s), PIN(w.lob_i, w.lob_s), mn, SZ, CTRI);
    GEMM(mn, w.wo, PIN(w.bo_i, w.bo_s), SZ, 128, 128, 0, z, nullptr, t1, 1);
  }

  // ================= Triangle attention (starting node) =================
  LN(z, PIN(69, 88), PIN(70, 87), zn, SZ, CZ);
  GEMM(zn, wq_ts, nullptr, SZ, 128, 128, 0, nullptr, qb, nullptr, 0);
  GEMM(zn, wk_ts, nullptr, SZ, 128, 128, 0, nullptr, kb, nullptr, 0);
  GEMM(zn, wv_ts, nullptr, SZ, 128, 128, 0, nullptr, vb, nullptr, 0);
  GEMM(zn, wg_ts, PIN(75, 85), SZ, 128, 128, 2, nullptr, gb, nullptr, 0);
  GEMM(zn, wb_ts, nullptr, SZ, 4, 128, 0, bia, nullptr, nullptr, 0);
  {
    AttnP p; p.q = qb; p.k = kb; p.v = vb; p.bias = bia; p.out = ato;
    p.q_bs = (long)Ll * HT * DHc; p.q_rs = HT * DHc;
    p.b_rs = (long)Ll * HT; p.b_cs = HT;
    p.Q = Ll; p.S = Ll; p.scale = SCALE;
    attn_kernel<<<dim3(Ll / 16, HT, Ll), 32, 0, stream>>>(p);
  }
  mul_bf_f32_kernel<<<EW(SZ * 128), 256, 0, stream>>>(gb, ato, go, SZ * 128);
  GEMM(go, wo_ts, PIN(77, 86), SZ, 128, 128, 0, z, nullptr, nullptr, 1);

  // ================= Triangle attention (ending node: on z^T) =================
  LN(z, PIN(79, 46), PIN(80, 45), zn, SZ, CZ);
  transpose_bf_kernel<<<EW(SZ * CZ), 256, 0, stream>>>(zn, znT, Ll, CZ);
  GEMM(znT, wq_te, nullptr, SZ, 128, 128, 0, nullptr, qb, nullptr, 0);
  GEMM(znT, wk_te, nullptr, SZ, 128, 128, 0, nullptr, kb, nullptr, 0);
  GEMM(znT, wv_te, nullptr, SZ, 128, 128, 0, nullptr, vb, nullptr, 0);
  GEMM(znT, wg_te, PIN(85, 43), SZ, 128, 128, 2, nullptr, gb, nullptr, 0);
  GEMM(znT, wb_te, nullptr, SZ, 4, 128, 0, bia, nullptr, nullptr, 0);
  {
    AttnP p; p.q = qb; p.k = kb; p.v = vb; p.bias = bia; p.out = ato;
    p.q_bs = (long)Ll * HT * DHc; p.q_rs = HT * DHc;
    p.b_rs = (long)Ll * HT; p.b_cs = HT;
    p.Q = Ll; p.S = Ll; p.scale = SCALE;
    attn_kernel<<<dim3(Ll / 16, HT, Ll), 32, 0, stream>>>(p);
  }
  mul_bf_f32_kernel<<<EW(SZ * 128), 256, 0, stream>>>(gb, ato, go, SZ * 128);
  GEMM(go, wo_te, PIN(87, 44), SZ, 128, 128, 0, t2, nullptr, nullptr, 0);
  addT_kernel<<<EW(SZ * CZ), 256, 0, stream>>>(z, t2, Ll, CZ);

  // ================= Pair transition =================
  LN(z, PIN(89, 40), PIN(90, 39), zn, SZ, CZ);
  GEMM(zn, w1_p, PIN(92, 37), SZ, 512, 128, 1, nullptr, hid, nullptr, 0);
  GEMM(hid, w2_p, PIN(94, 38), SZ, 128, 512, 0, z, nullptr, nullptr, 1);
}
